// HyperbolicContrastiveLoss_75136157876227
// MI455X (gfx1250) — matrix operand order repaired and checked by reading, not compiled, not run on hardware
//
#include <hip/hip_runtime.h>
#include <math.h>

// CDNA5 wave32 WMMA types
typedef __attribute__((ext_vector_type(16))) __bf16 v16bf;
typedef __attribute__((ext_vector_type(8)))  __bf16 v8bf;
typedef __attribute__((ext_vector_type(8)))  float  v8f;

#define EPSF 1e-10f

// ---- bf16 <-> f32 bit helpers (round-to-nearest-even) ----
__device__ inline unsigned short f2bf_bits(float x) {
    union { float f; unsigned u; } v; v.f = x;
    unsigned r = v.u + 0x7FFFu + ((v.u >> 16) & 1u);
    return (unsigned short)(r >> 16);
}
__device__ inline float bf_bits2f(unsigned short s) {
    union { float f; unsigned u; } v; v.u = ((unsigned)s) << 16;
    return v.f;
}

// ---- WMMA fragment loaders, straight from L2 (32MB working set << 192MB L2) ----
// A-matrix 16x32 bf16 layout (ISA 7.12.2): lane L holds row M=L%16;
// lanes 0-15: K=[0..7] in V0..3, K=[16..23] in V4..7; lanes 16-31: +8.
__device__ inline v16bf load_a_frag(const unsigned short* __restrict__ Abase,
                                    int D, int m, int k0, int lane) {
    int kh = (lane & 16) ? 8 : 0;
    const unsigned short* p = Abase + (size_t)(m + (lane & 15)) * D + k0 + kh;
    v8bf lo = *reinterpret_cast<const v8bf*>(p);        // K = k0+kh .. +7
    v8bf hi = *reinterpret_cast<const v8bf*>(p + 16);   // K = k0+16+kh .. +7
    return __builtin_shufflevector(lo, hi, 0,1,2,3,4,5,6,7,8,9,10,11,12,13,14,15);
}
// B-matrix 32x16 bf16: lane L holds col N=L%16; lanes 0-15: K=[0..7],[16..23],
// lanes 16-31: +8 (same K order as A). B(k,n) = text[n][k].
__device__ inline v16bf load_b_frag(const unsigned short* __restrict__ Bbase,
                                    int D, int n, int k0, int lane) {
    int ko = (lane & 16) ? 8 : 0;
    const unsigned short* p = Bbase + (size_t)(n + (lane & 15)) * D + k0 + ko;
    v8bf lo = *reinterpret_cast<const v8bf*>(p);        // K = k0+ko .. +7
    v8bf hi = *reinterpret_cast<const v8bf*>(p + 16);   // K = k0+16+ko .. +7
    return __builtin_shufflevector(lo, hi, 0,1,2,3,4,5,6,7,8,9,10,11,12,13,14,15);
}

__device__ inline v8f wmma_bf16(v16bf a, v16bf b, v8f c) {
    // (neg_a, A, neg_b, B, c_mod, C, reuse_a, reuse_b)
    return __builtin_amdgcn_wmma_f32_16x16x32_bf16(false, a, false, b,
                                                   (short)0, c, false, false);
}

// ---- Kernel 0: zero the global accumulators (harness poisons ws once) ----
__global__ void zero_kernel(float* rowsum, float* colsum, int N) {
    int i = blockIdx.x * blockDim.x + threadIdx.x;
    if (i < N) { rowsum[i] = 0.f; colsum[i] = 0.f; }
}

// ---- Kernel 1: fp32 -> bf16 hi/lo split + squared norms (fp32-exact) ----
__global__ __launch_bounds__(128) void prep_kernel(
    const float* __restrict__ A, const float* __restrict__ B,
    unsigned short* __restrict__ Ah, unsigned short* __restrict__ Al,
    unsigned short* __restrict__ Bh, unsigned short* __restrict__ Bl,
    float* __restrict__ a2, float* __restrict__ b2, int N, int D)
{
    __shared__ float red[128];
    int b = blockIdx.x;
    bool isB = b >= N;
    int row = isB ? (b - N) : b;
    const float* src = isB ? B : A;
    unsigned short* dh = isB ? Bh : Ah;
    unsigned short* dl = isB ? Bl : Al;
    float* norm = isB ? b2 : a2;

    float s = 0.f;
    for (int k = threadIdx.x; k < D; k += blockDim.x) {
        float x = src[(size_t)row * D + k];
        unsigned short h = f2bf_bits(x);
        float hf = bf_bits2f(h);
        unsigned short l = f2bf_bits(x - hf);
        dh[(size_t)row * D + k] = h;
        dl[(size_t)row * D + k] = l;
        s += x * x;
    }
    red[threadIdx.x] = s; __syncthreads();
    for (int off = 64; off > 0; off >>= 1) {
        if (threadIdx.x < off) red[threadIdx.x] += red[threadIdx.x + off];
        __syncthreads();
    }
    if (threadIdx.x == 0) norm[row] = red[0];
}

// ---- Kernel 2: fused 128x128 tile: WMMA dot products + hyperbolic transform
//      + streaming row/col exp-sums + label-entry capture. exp(-arccosh(x)) =
//      1/(x+sqrt(x^2-1)) -> no transcendentals in the N^2 loop.
__global__ __launch_bounds__(256) void sim_tile_kernel(
    const unsigned short* __restrict__ Ah, const unsigned short* __restrict__ Al,
    const unsigned short* __restrict__ Bh, const unsigned short* __restrict__ Bl,
    const float* __restrict__ a2, const float* __restrict__ b2,
    const int* __restrict__ labels,
    float* __restrict__ rowsum, float* __restrict__ colsum,
    float* __restrict__ diagA, float* __restrict__ diagB,
    int N, int D)
{
    __shared__ float rowacc[128];
    __shared__ float colacc[128];
    int tid = threadIdx.x;
    if (tid < 128) rowacc[tid] = 0.f; else colacc[tid - 128] = 0.f;
    __syncthreads();

    const int i0 = blockIdx.y * 128, j0 = blockIdx.x * 128;
    const int w = tid >> 5, lane = tid & 31;
    const int wm = w >> 1, wn = w & 1;           // 4x2 wave grid -> 32x64 per wave
    const int mBase = i0 + wm * 32;
    const int nBase = j0 + wn * 64;

    v8f acc[2][4];
    const v8f vzero = {0.f,0.f,0.f,0.f,0.f,0.f,0.f,0.f};
#pragma unroll
    for (int mt = 0; mt < 2; mt++)
#pragma unroll
        for (int nt = 0; nt < 4; nt++) acc[mt][nt] = vzero;

    // K loop: bf16 hi/lo split => dot = hi*hi + hi*lo + lo*hi (fp32-class accuracy)
    for (int k0 = 0; k0 < D; k0 += 32) {
        v16bf ah0 = load_a_frag(Ah, D, mBase,      k0, lane);
        v16bf ah1 = load_a_frag(Ah, D, mBase + 16, k0, lane);
        v16bf al0 = load_a_frag(Al, D, mBase,      k0, lane);
        v16bf al1 = load_a_frag(Al, D, mBase + 16, k0, lane);
#pragma unroll
        for (int nt = 0; nt < 4; nt++) {
            v16bf bh = load_b_frag(Bh, D, nBase + nt * 16, k0, lane);
            v16bf bl = load_b_frag(Bl, D, nBase + nt * 16, k0, lane);
            acc[0][nt] = wmma_bf16(ah0, bh, acc[0][nt]);
            acc[0][nt] = wmma_bf16(ah0, bl, acc[0][nt]);
            acc[0][nt] = wmma_bf16(al0, bh, acc[0][nt]);
            acc[1][nt] = wmma_bf16(ah1, bh, acc[1][nt]);
            acc[1][nt] = wmma_bf16(ah1, bl, acc[1][nt]);
            acc[1][nt] = wmma_bf16(al1, bh, acc[1][nt]);
        }
    }

    // Elementwise stage. C/D layout: VGPR v, lane L -> (M = v + (L>=16)*8, N = L&15)
    const int half = lane >> 4, nIdx = lane & 15;
    float one_m_b[4], bn2_[4]; int gn_[4], labn[4];
#pragma unroll
    for (int nt = 0; nt < 4; nt++) {
        int gn = nBase + nt * 16 + nIdx;
        gn_[nt] = gn;
        float bn2 = b2[gn]; bn2_[nt] = bn2;
        one_m_b[nt] = 1.f - fminf(bn2, 1.f - EPSF);
        labn[nt] = labels[gn];
    }
    float colp[4] = {0.f, 0.f, 0.f, 0.f};
#pragma unroll
    for (int mt = 0; mt < 2; mt++) {
#pragma unroll
        for (int v = 0; v < 8; v++) {
            int gm = mBase + mt * 16 + half * 8 + v;
            float am2 = a2[gm];
            int labm = labels[gm];
            float one_m_a = 1.f - fminf(am2, 1.f - EPSF);
            float rowp = 0.f;
#pragma unroll
            for (int nt = 0; nt < 4; nt++) {
                float dot  = acc[mt][nt][v];
                float diff = am2 + bn2_[nt] - 2.f * dot;        // ||a-b||^2
                float denom = fmaxf(one_m_a * one_m_b[nt], EPSF);
                float arg = fmaxf(1.f + 2.f * diff / denom, 1.f + EPSF);
                float p = 1.f / (arg + sqrtf(arg * arg - 1.f)); // exp(-arccosh(arg))
                rowp += p; colp[nt] += p;
                if (gn_[nt] == labm) diagA[gm] = p;             // p(i, label[i])
                if (gm == labn[nt]) diagB[gn_[nt]] = p;         // p(label[j], j)
            }
            // reduce over the 16 lanes of this half (all same row gm)
            rowp += __shfl_xor(rowp, 1, 32);
            rowp += __shfl_xor(rowp, 2, 32);
            rowp += __shfl_xor(rowp, 4, 32);
            rowp += __shfl_xor(rowp, 8, 32);
            if (nIdx == 0) atomicAdd(&rowacc[gm - i0], rowp);
        }
    }
#pragma unroll
    for (int nt = 0; nt < 4; nt++) {
        float c = colp[nt] + __shfl_xor(colp[nt], 16, 32);      // combine M halves
        if (half == 0) atomicAdd(&colacc[gn_[nt] - j0], c);
    }
    __syncthreads();
    if (tid < 128) atomicAdd(&rowsum[i0 + tid], rowacc[tid]);
    else           atomicAdd(&colsum[j0 + tid - 128], colacc[tid - 128]);
}

// ---- Kernel 3: loss = mean_i[ 0.5*(log rs_i - log pA_i + log cs_i - log pB_i) ]
__global__ __launch_bounds__(256) void reduce_kernel(
    const float* __restrict__ rowsum, const float* __restrict__ colsum,
    const float* __restrict__ diagA, const float* __restrict__ diagB,
    float* __restrict__ out, int N)
{
    __shared__ float red[256];
    float s = 0.f;
    for (int i = threadIdx.x; i < N; i += 256) {
        s += 0.5f * (logf(rowsum[i]) - logf(diagA[i]) +
                     logf(colsum[i]) - logf(diagB[i]));
    }
    red[threadIdx.x] = s; __syncthreads();
    for (int off = 128; off > 0; off >>= 1) {
        if (threadIdx.x < off) red[threadIdx.x] += red[threadIdx.x + off];
        __syncthreads();
    }
    if (threadIdx.x == 0) out[0] = red[0] / (float)N;
}

extern "C" void kernel_launch(void* const* d_in, const int* in_sizes, int n_in,
                              void* d_out, int out_size, void* d_ws, size_t ws_size,
                              hipStream_t stream)
{
    const float* A      = (const float*)d_in[0];   // audio_features [N,D] f32
    const float* B      = (const float*)d_in[1];   // text_features  [N,D] f32
    const int*   labels = (const int*)d_in[2];     // [N] i32

    const int N = in_sizes[2];
    const int D = in_sizes[0] / N;                 // N=8192 (mult of 128), D=512 (mult of 32)

    // workspace carving (bf16 hi/lo: 4 * N*D*2 B = 32MB; 6 float[N] arrays = 192KB)
    char* ws = (char*)d_ws;
    size_t ND = (size_t)N * D;
    unsigned short* Ah = (unsigned short*)ws; ws += ND * 2;
    unsigned short* Al = (unsigned short*)ws; ws += ND * 2;
    unsigned short* Bh = (unsigned short*)ws; ws += ND * 2;
    unsigned short* Bl = (unsigned short*)ws; ws += ND * 2;
    float* a2     = (float*)ws; ws += (size_t)N * 4;
    float* b2     = (float*)ws; ws += (size_t)N * 4;
    float* rowsum = (float*)ws; ws += (size_t)N * 4;
    float* colsum = (float*)ws; ws += (size_t)N * 4;
    float* diagA  = (float*)ws; ws += (size_t)N * 4;
    float* diagB  = (float*)ws; ws += (size_t)N * 4;

    zero_kernel<<<(N + 255) / 256, 256, 0, stream>>>(rowsum, colsum, N);
    prep_kernel<<<2 * N, 128, 0, stream>>>(A, B, Ah, Al, Bh, Bl, a2, b2, N, D);
    dim3 grid(N / 128, N / 128);
    sim_tile_kernel<<<grid, 256, 0, stream>>>(Ah, Al, Bh, Bl, a2, b2, labels,
                                              rowsum, colsum, diagA, diagB, N, D);
    reduce_kernel<<<1, 256, 0, stream>>>(rowsum, colsum, diagA, diagB,
                                         (float*)d_out, N);
}
